// GCNdesign_simple_18519898981008
// MI455X (gfx1250) — matrix operand order repaired and checked by reading, not compiled, not run on hardware
//
#include <hip/hip_runtime.h>
#include <cstdint>

// ---------------------------------------------------------------------------
// CDNA5 (gfx1250) WMMA implementation of the GCNdesign forward pass.
//  * all dense layers -> v_wmma_f32_16x16x32_bf16 (bf16 in, f32 accum)
//  * LayerNorm+ReLU fused into GEMM A-tile staging (only per-row mu/rstd kept)
//  * the 230MB message tensor m = concat(node_i, node_j, edge) is NEVER
//    materialized: e1 GEMMs gather it on the fly; the segment select is
//    uniform per K-tile (dn, 2dn, 2dn+DE all multiples of 32) -> scalar branch
//  * LDS tiles in WMMA fragment order -> 2x ds_load_b128 per fragment,
//    packed u32 staging stores (native v_cvt bf16 pack), double-buffered
//    (1 barrier / K-step)
// ---------------------------------------------------------------------------

typedef __attribute__((ext_vector_type(16))) __bf16          v16bf;
typedef __attribute__((ext_vector_type(2)))  __bf16          v2bf;
typedef __attribute__((ext_vector_type(8)))  float           v8f;

#define EPSF 1e-5f

// problem constants (compile-time so index divisions strength-reduce)
constexpr int Bv = 2, Lv = 4096, Kv = 20;
constexpr int Nn = Bv * Lv;                    // 8192 node tokens
constexpr long long Ee = (long long)Nn * Kv;   // 163840 edge tokens
constexpr int DN = 128, DE = 32, DI = 32, DOUTv = 20, DEIN = 36;

// pack two f32 -> u32 of two bf16 (RNE); lets clang emit native cvt_pk
__device__ __forceinline__ unsigned pack_bf16(float v0, float v1) {
    v2bf p;
    p[0] = (__bf16)v0;
    p[1] = (__bf16)v1;
    return __builtin_bit_cast(unsigned, p);
}

// virtual-A context for gather GEMMs (A row = concat(node_i, node_j, edge))
struct GatherCtx {
    const float* nodef;   // (Nn, dn)
    const float* edgef;   // (Ee, DE)
    const int*   njoff;   // (Ee) precomputed (b*Lv + adj[e]) * dn
    int dn;
};

// ---------------------------------------------------------------------------
// WMMA GEMM:  C[T x dout] = LN?(A)[T x din] @ W[din x dout] + bias (+ addsrc)
//   8 wave32s / block, block tile 128x16, K-step 32 (bf16), double-buffered.
// ---------------------------------------------------------------------------
#define GEMM_MT 128
#define GEMM_NT 16
#define GEMM_KT 32

template<bool GATHER, bool LNRELU, bool HASADD>
__global__ __launch_bounds__(256)
void gemm_wmma_kernel(const float* __restrict__ A, const float* __restrict__ W,
                      const float* __restrict__ bias, const float* __restrict__ addsrc,
                      const float* __restrict__ mu, const float* __restrict__ rstd,
                      const float* __restrict__ lng, const float* __restrict__ lnb,
                      float* __restrict__ C, int T, int din, int dout,
                      GatherCtx g)
{
    // double-buffered tiles, u32-packed bf16 pairs, fragment-ordered:
    //  A: row-major rows, K contiguous [0..31]
    //  W: column-major
    __shared__ __align__(16) unsigned sAu[2 * GEMM_MT * GEMM_KT / 2];
    __shared__ __align__(16) unsigned sWu[2 * GEMM_NT * GEMM_KT / 2];

    const int tid  = threadIdx.x;
    const int wave = tid >> 5;
    const int lane = tid & 31;
    const int row0 = blockIdx.x * GEMM_MT;
    const int col0 = blockIdx.y * GEMM_NT;

    // per-thread fixed staging coordinates
    const int apc = tid & 15;              // A pair-column (k = 2*apc)
    const int ac  = apc * 2;
    const int ap2 = apc;
    const int wn  = tid & 15;              // W column
    const int wk  = (tid >> 4) * 2;        // W k-pair

    float a0[8], a1[8];                    // staged A pairs (8 rows / thread)
    float w0, w1;                          // staged W pair

    // ---- phase 1: global loads (+ fused LN/ReLU) into registers ----
    auto stage_load = [&](int k0) {
        #pragma unroll
        for (int i = 0; i < 8; ++i) {
            const int row = row0 + (tid >> 4) + 16 * i;
            float v0 = 0.f, v1 = 0.f;
            bool ok = (row < T);
            if constexpr (!GATHER) ok = ok && (k0 + ac < din);
            if (ok) {
                if constexpr (GATHER) {
                    // segment select is uniform over the whole K-tile
                    const float2* src;
                    if (k0 < g.dn) {
                        long long bl = (long long)row / Kv;       // b*L + l
                        src = (const float2*)(g.nodef + bl * g.dn + k0 + ac);
                    } else if (k0 < 2 * g.dn) {
                        src = (const float2*)(g.nodef + g.njoff[row] + (k0 - g.dn) + ac);
                    } else {
                        src = (const float2*)(g.edgef + (size_t)row * DE + (k0 - 2 * g.dn) + ac);
                    }
                    float2 t2 = *src;
                    v0 = t2.x; v1 = t2.y;
                } else {
                    float2 t2 = *(const float2*)(A + (size_t)row * din + k0 + ac);
                    v0 = t2.x; v1 = t2.y;
                }
                if constexpr (LNRELU) {
                    const float mr = mu[row], rr = rstd[row];
                    const float2 gg = *(const float2*)(lng + k0 + ac);
                    const float2 bb = *(const float2*)(lnb + k0 + ac);
                    v0 = (v0 - mr) * rr * gg.x + bb.x;
                    v1 = (v1 - mr) * rr * gg.y + bb.y;
                    v0 = v0 > 0.f ? v0 : 0.f;
                    v1 = v1 > 0.f ? v1 : 0.f;
                }
            }
            a0[i] = v0; a1[i] = v1;
        }
        {
            const int kk = k0 + wk, cc = col0 + wn;
            w0 = w1 = 0.f;
            if (cc < dout && kk < din) {           // din even -> pair all-in
                w0 = W[(size_t)kk * dout + cc];
                w1 = W[(size_t)(kk + 1) * dout + cc];
            }
        }
    };
    // ---- phase 2: packed u32 stores into LDS tile `sel` ----
    auto stage_store = [&](int sel) {
        unsigned* sA = sAu + sel * (GEMM_MT * GEMM_KT / 2);
        unsigned* sW = sWu + sel * (GEMM_NT * GEMM_KT / 2);
        #pragma unroll
        for (int i = 0; i < 8; ++i) {
            const int r = (tid >> 4) + 16 * i;
            sA[r * 16 + ap2] = pack_bf16(a0[i], a1[i]);
        }
        sW[wn * 16 + (wk >> 1)] = pack_bf16(w0, w1);
    };

    v8f acc = {0.f, 0.f, 0.f, 0.f, 0.f, 0.f, 0.f, 0.f};
    const int nk = (din + GEMM_KT - 1) / GEMM_KT;

    stage_load(0);
    stage_store(0);
    __syncthreads();

    int cur = 0;
    for (int t = 0; t < nk; ++t) {
        if (t + 1 < nk) stage_load((t + 1) * GEMM_KT);   // overlap with WMMA

        {
            struct U128x2 { uint4 x, y; };
            const unsigned short* sA16 =
                (const unsigned short*)(sAu + cur * (GEMM_MT * GEMM_KT / 2));
            const unsigned short* sW16 =
                (const unsigned short*)(sWu + cur * (GEMM_NT * GEMM_KT / 2));
            const int m = lane & 15;
            const unsigned short* ap =
                &sA16[(wave * 16 + m) * GEMM_KT + ((lane < 16) ? 0 : 16)];
            U128x2 apk; apk.x = *(const uint4*)ap; apk.y = *(const uint4*)(ap + 8);
            const int kb = (lane < 16) ? 0 : 16;
            const unsigned short* bp = &sW16[m * GEMM_KT + kb];
            U128x2 bpk; bpk.x = *(const uint4*)bp; bpk.y = *(const uint4*)(bp + 8);
            v16bf av = __builtin_bit_cast(v16bf, apk);
            v16bf bv = __builtin_bit_cast(v16bf, bpk);
            acc = __builtin_amdgcn_wmma_f32_16x16x32_bf16(
                false, av, false, bv, (short)0, acc, false, false);
        }

        if (t + 1 < nk) stage_store(cur ^ 1);
        __syncthreads();
        cur ^= 1;
    }

    // ---- epilogue: bias + optional residual ----
    const int col = col0 + (lane & 15);
    if (col < dout) {
        const float bb    = bias[col];
        const int   rbase = row0 + wave * 16 + ((lane < 16) ? 0 : 8);
        #pragma unroll
        for (int j = 0; j < 8; ++j) {
            int row = rbase + j;
            if (row < T) {
                float v = acc[j] + bb;
                if constexpr (HASADD) v += addsrc[(size_t)row * dout + col];
                C[(size_t)row * dout + col] = v;
            }
        }
    }
}

// ---------------------------------------------------------------------------
// njoff[e] = (b*Lv + adj[e]) * dn   (per-layer neighbor row offsets)
// ---------------------------------------------------------------------------
__global__ __launch_bounds__(256)
void prep_nj_kernel(const int* __restrict__ adj, int dn, int* __restrict__ njoff)
{
    long long e = (long long)blockIdx.x * blockDim.x + threadIdx.x;
    if (e >= Ee) return;
    long long bb = (e / Kv) / Lv;
    njoff[e] = (int)((bb * Lv + adj[e]) * dn);
}

// ---------------------------------------------------------------------------
// Per-row LayerNorm statistics (one wave32 per row)
// ---------------------------------------------------------------------------
__global__ __launch_bounds__(256)
void rowstats_kernel(const float* __restrict__ x, int T, int d,
                     float* __restrict__ mu, float* __restrict__ rstd)
{
    const int wave = threadIdx.x >> 5, lane = threadIdx.x & 31;
    const int row  = blockIdx.x * 8 + wave;
    if (row >= T) return;
    const float* xr = x + (size_t)row * d;
    float s = 0.f, q = 0.f;
    for (int c = lane; c < d; c += 32) { float v = xr[c]; s += v; q += v * v; }
    #pragma unroll
    for (int off = 16; off > 0; off >>= 1) {
        s += __shfl_xor(s, off, 32);
        q += __shfl_xor(q, off, 32);
    }
    if (lane == 0) {
        float m   = s / (float)d;
        float var = q / (float)d - m * m;
        var = var < 0.f ? 0.f : var;
        mu[row]   = m;
        rstd[row] = rsqrtf(var + EPSF);
    }
}

// ---------------------------------------------------------------------------
// Row stats over the VIRTUAL m row (never materializes m); wave32 per token
// ---------------------------------------------------------------------------
__global__ __launch_bounds__(256)
void rowstats_gather_kernel(const float* __restrict__ nodef,
                            const float* __restrict__ edgef,
                            const int* __restrict__ njoff, int dn,
                            int T, float* __restrict__ mu, float* __restrict__ rstd)
{
    const int wave = threadIdx.x >> 5, lane = threadIdx.x & 31;
    const int row  = blockIdx.x * 8 + wave;
    if (row >= T) return;
    const long long bl = (long long)row / Kv;
    const float* ni = nodef + bl * dn;
    const float* nj = nodef + njoff[row];
    const float* ef = edgef + (size_t)row * DE;
    const int dm = 2 * dn + DE;
    float s = 0.f, q = 0.f;
    for (int c = lane; c < dm; c += 32) {
        float v = (c < dn) ? ni[c] : ((c < 2 * dn) ? nj[c - dn] : ef[c - 2 * dn]);
        s += v; q += v * v;
    }
    #pragma unroll
    for (int off = 16; off > 0; off >>= 1) {
        s += __shfl_xor(s, off, 32);
        q += __shfl_xor(q, off, 32);
    }
    if (lane == 0) {
        float m   = s / (float)dm;
        float var = q / (float)dm - m * m;
        var = var < 0.f ? 0.f : var;
        mu[row]   = m;
        rstd[row] = rsqrtf(var + EPSF);
    }
}

// ---------------------------------------------------------------------------
// Standalone LayerNorm + ReLU, one wave32 per row
// ---------------------------------------------------------------------------
__global__ __launch_bounds__(256)
void lnrelu_kernel(const float* __restrict__ x, const float* __restrict__ g,
                   const float* __restrict__ b, float* __restrict__ y,
                   int T, int d)
{
    const int wave = threadIdx.x >> 5, lane = threadIdx.x & 31;
    const int row  = blockIdx.x * 8 + wave;
    if (row >= T) return;
    const float* xr = x + (size_t)row * d;
    float s = 0.f, q = 0.f;
    for (int c = lane; c < d; c += 32) { float v = xr[c]; s += v; q += v * v; }
    #pragma unroll
    for (int off = 16; off > 0; off >>= 1) {
        s += __shfl_xor(s, off, 32);
        q += __shfl_xor(q, off, 32);
    }
    float m   = s / (float)d;
    float var = q / (float)d - m * m;
    var = var < 0.f ? 0.f : var;
    float rs  = rsqrtf(var + EPSF);
    float* yr = y + (size_t)row * d;
    for (int c = lane; c < d; c += 32) {
        float v = (xr[c] - m) * rs * g[c] + b[c];
        yr[c] = v > 0.f ? v : 0.f;
    }
}

// ---------------------------------------------------------------------------
// 1-D conv, kernel 9, pad 4: node (B,L,6) -> y (B,L,128)
// ---------------------------------------------------------------------------
__global__ __launch_bounds__(256)
void conv1d_kernel(const float* __restrict__ node, const float* __restrict__ w,
                   const float* __restrict__ b, float* __restrict__ y,
                   int cin, int cout)
{
    long long idx   = (long long)blockIdx.x * blockDim.x + threadIdx.x;
    long long total = (long long)Nn * cout;
    if (idx >= total) return;
    int       o  = (int)(idx % cout);
    long long t  = idx / cout;
    int       l  = (int)(t % Lv);
    long long bb = t / Lv;
    float acc = b[o];
    for (int ww = 0; ww < 9; ++ww) {
        int ll = l + ww - 4;
        if (ll < 0 || ll >= Lv) continue;
        const float* np = node + ((size_t)(bb * Lv + ll)) * cin;
        const float* wp = w + ((size_t)ww * cin) * cout + o;
        for (int ci = 0; ci < cin; ++ci) acc += np[ci] * wp[(size_t)ci * cout];
    }
    y[(size_t)t * cout + o] = acc;
}

// ---------------------------------------------------------------------------
// Row gather: dst[e,:] = src[idx[e],:]
// ---------------------------------------------------------------------------
__global__ __launch_bounds__(256)
void gather_rows_kernel(const float* __restrict__ src, const int* __restrict__ idx,
                        float* __restrict__ dst, long long n, int d)
{
    long long i = (long long)blockIdx.x * blockDim.x + threadIdx.x;
    if (i >= n * (long long)d) return;
    int       c = (int)(i % d);
    long long e = i / d;
    dst[i] = src[(size_t)idx[e] * d + c];
}

// ---------------------------------------------------------------------------
// h[t,:] = concat(node[t,:dn], mean_k edge[t,k,:])   (mean fused)
// ---------------------------------------------------------------------------
__global__ __launch_bounds__(256)
void concat_node_agg_kernel(const float* __restrict__ nodef, int dn,
                            const float* __restrict__ edgef,
                            float* __restrict__ h)
{
    const int d = dn + DE;
    long long total = (long long)Nn * d;
    long long i = (long long)blockIdx.x * blockDim.x + threadIdx.x;
    if (i >= total) return;
    int       c = (int)(i % d);
    long long t = i / d;
    if (c < dn) {
        h[i] = nodef[t * (size_t)dn + c];
    } else {
        float s = 0.f;
        const float* ep = edgef + (t * Kv) * (size_t)DE + (c - dn);
        #pragma unroll
        for (int k = 0; k < Kv; ++k) s += ep[(size_t)k * DE];
        h[i] = s * (1.f / (float)Kv);
    }
}

// ---------------------------------------------------------------------------
// out[t,:] = concat(X[t,:d1], Y[t,:d2])
// ---------------------------------------------------------------------------
__global__ __launch_bounds__(256)
void concat2_kernel(const float* __restrict__ X, int d1,
                    const float* __restrict__ Y, int d2,
                    float* __restrict__ out, long long T)
{
    const int d = d1 + d2;
    long long i = (long long)blockIdx.x * blockDim.x + threadIdx.x;
    if (i >= T * (long long)d) return;
    int       c = (int)(i % d);
    long long t = i / d;
    out[i] = (c < d1) ? X[t * (size_t)d1 + c] : Y[t * (size_t)d2 + (c - d1)];
}

__global__ __launch_bounds__(256)
void copy_kernel(const float* __restrict__ src, float* __restrict__ dst, long long n)
{
    long long i = (long long)blockIdx.x * blockDim.x + threadIdx.x;
    if (i < n) dst[i] = src[i];
}

// ===========================================================================
// Host-side orchestration
// ===========================================================================
namespace {

struct LinP { const float* w; const float* b; };
struct LNP  { const float* g; const float* b; };
struct ResP { LNP ln1; LinP fc1; LNP ln2; LinP fc2; LinP sh; bool has_short; int din, dout; };
struct RgcP { ResP e1, e2; LNP eln; ResP n1, n2; LNP nln; };

struct Reader {
    void* const* d; int i;
    const float* f() { return (const float*)d[i++]; }
    const int*   s() { return (const int*)d[i++]; }
};

static LNP  readLN (Reader& R) { LNP p; p.g = R.f(); p.b = R.f(); return p; }
static LinP readLin(Reader& R) { LinP p; p.w = R.f(); p.b = R.f(); return p; }
static ResP readRes(Reader& R, int din, int dout) {
    ResP r; r.din = din; r.dout = dout;
    r.ln1 = readLN(R); r.fc1 = readLin(R);
    r.ln2 = readLN(R); r.fc2 = readLin(R);
    r.has_short = (din != dout);
    if (r.has_short) r.sh = readLin(R); else r.sh = LinP{nullptr, nullptr};
    return r;
}
static RgcP readRgc(Reader& R, int dn, int de) {
    RgcP g;
    g.e1 = readRes(R, 2 * dn + de, de);
    g.e2 = readRes(R, de, de);
    g.eln = readLN(R);
    g.n1 = readRes(R, dn + de, dn);
    g.n2 = readRes(R, dn, dn);
    g.nln = readLN(R);
    return g;
}

struct WsBuf {
    float *e36;
    float *e_a, *e_b, *etmp, *ebuf;
    float *mu, *rstd;
    float *n_a, *n_b, *ntmp, *hbuf, *xbuf;
    float *idxb;
    int   *njoff;
};

static inline unsigned gblk(long long n, int b = 256) {
    return (unsigned)((n + b - 1) / b);
}

static void launch_gemm(hipStream_t s, const float* A, int T, int din, int dout,
                        const LinP& lin, const float* addsrc,
                        const float* mu, const float* rstd, const LNP* ln,
                        bool lnrelu, float* C, const GatherCtx* gp)
{
    dim3 grid((T + GEMM_MT - 1) / GEMM_MT, (dout + GEMM_NT - 1) / GEMM_NT);
    GatherCtx gc = gp ? *gp : GatherCtx{nullptr, nullptr, nullptr, 0};
    const float* lg = ln ? ln->g : nullptr;
    const float* lb = ln ? ln->b : nullptr;
    const bool gather = (gp != nullptr);
    const bool hasadd = (addsrc != nullptr);
#define GEMM_CALL(G, LR, HA)                                                   \
    gemm_wmma_kernel<G, LR, HA><<<grid, 256, 0, s>>>(                          \
        A, lin.w, lin.b, addsrc, mu, rstd, lg, lb, C, T, din, dout, gc)
    if (gather) {
        if (lnrelu) { if (hasadd) GEMM_CALL(true,  true,  true);  else GEMM_CALL(true,  true,  false); }
        else        { if (hasadd) GEMM_CALL(true,  false, true);  else GEMM_CALL(true,  false, false); }
    } else {
        if (lnrelu) { if (hasadd) GEMM_CALL(false, true,  true);  else GEMM_CALL(false, true,  false); }
        else        { if (hasadd) GEMM_CALL(false, false, true);  else GEMM_CALL(false, false, false); }
    }
#undef GEMM_CALL
}

static void launch_rowstats(hipStream_t s, const float* x, int T, int d,
                            float* mu, float* rstd)
{
    rowstats_kernel<<<dim3((T + 7) / 8), 256, 0, s>>>(x, T, d, mu, rstd);
}

static void launch_lnrelu(hipStream_t s, const float* x, const LNP& ln,
                          float* y, int T, int d)
{
    lnrelu_kernel<<<dim3((T + 7) / 8), 256, 0, s>>>(x, ln.g, ln.b, y, T, d);
}

// res block with materialized input x
static void run_res(hipStream_t s, const ResP& r, const float* x, float* out,
                    float* tmp, float* mu, float* rstd, int T)
{
    launch_rowstats(s, x, T, r.din, mu, rstd);
    launch_gemm(s, x, T, r.din, r.dout, r.fc1, nullptr, mu, rstd, &r.ln1, true, tmp, nullptr);
    launch_rowstats(s, tmp, T, r.dout, mu, rstd);
    if (r.has_short) {
        launch_gemm(s, x,   T, r.din,  r.dout, r.sh,  nullptr, nullptr, nullptr, nullptr, false, out, nullptr);
        launch_gemm(s, tmp, T, r.dout, r.dout, r.fc2, out,     mu, rstd, &r.ln2, true, out, nullptr);
    } else {
        launch_gemm(s, tmp, T, r.dout, r.dout, r.fc2, x,       mu, rstd, &r.ln2, true, out, nullptr);
    }
}

// res block whose input is the VIRTUAL m tensor (gathered on the fly)
static void run_res_gather(hipStream_t s, const ResP& r, const GatherCtx& gc,
                           float* out, float* tmp, float* mu, float* rstd, int T)
{
    rowstats_gather_kernel<<<dim3((T + 7) / 8), 256, 0, s>>>(
        gc.nodef, gc.edgef, gc.njoff, gc.dn, T, mu, rstd);
    launch_gemm(s, nullptr, T, r.din, r.dout, r.fc1, nullptr, mu, rstd, &r.ln1, true, tmp, &gc);
    launch_rowstats(s, tmp, T, r.dout, mu, rstd);
    launch_gemm(s, nullptr, T, r.din,  r.dout, r.sh,  nullptr, nullptr, nullptr, nullptr, false, out, &gc);
    launch_gemm(s, tmp,     T, r.dout, r.dout, r.fc2, out,     mu, rstd, &r.ln2, true, out, nullptr);
}

static void run_rgc(hipStream_t s, const RgcP& g, int dn, const int* adj, WsBuf& w)
{
    prep_nj_kernel<<<dim3(gblk(Ee)), 256, 0, s>>>(adj, dn, w.njoff);
    GatherCtx gc{w.xbuf, w.ebuf, w.njoff, dn};
    // edge update (e1 gathers m on the fly)
    run_res_gather(s, g.e1, gc, w.e_a, w.etmp, w.mu, w.rstd, (int)Ee);
    run_res(s, g.e2, w.e_a, w.e_b, w.etmp, w.mu, w.rstd, (int)Ee);
    launch_lnrelu(s, w.e_b, g.eln, w.ebuf, (int)Ee, DE);
    // node update (mean-over-K fused into concat)
    concat_node_agg_kernel<<<dim3(gblk((long long)Nn * (dn + DE))), 256, 0, s>>>(
        w.xbuf, dn, w.ebuf, w.hbuf);
    run_res(s, g.n1, w.hbuf, w.n_a, w.ntmp, w.mu, w.rstd, Nn);
    run_res(s, g.n2, w.n_a,  w.n_b, w.ntmp, w.mu, w.rstd, Nn);
    launch_lnrelu(s, w.n_b, g.nln, w.xbuf, Nn, dn);
}

} // namespace

extern "C" void kernel_launch(void* const* d_in, const int* in_sizes, int n_in,
                              void* d_out, int out_size, void* d_ws, size_t ws_size,
                              hipStream_t stream)
{
    (void)in_sizes; (void)n_in; (void)out_size; (void)ws_size;

    // ---- inputs & params (reference dict insertion order) ----
    Reader R{d_in, 0};
    const float* node    = R.f();
    const float* edgemat = R.f();
    const int*   adjmat  = R.s();
    const int*   masked  = R.s();

    const float* conv_w = R.f();
    const float* conv_b = R.f();
    ResP nf_r1 = readRes(R, DN, DN);
    ResP nf_r2 = readRes(R, DN, DN);
    LNP  nf_ln = readLN(R);
    ResP ef_r1 = readRes(R, DEIN, DE);
    ResP ef_r2 = readRes(R, DE, DE);
    LNP  ef_ln = readLN(R);
    RgcP enc[4];
    for (int l = 0; l < 4; ++l) enc[l] = readRgc(R, DN, DE);
    const float* emb = R.f();
    ResP idx_r1 = readRes(R, DI, DI);
    LNP  idx_ln = readLN(R);
    RgcP dec[4];
    for (int l = 0; l < 4; ++l) dec[l] = readRgc(R, DN + DI, DE);
    LinP outp = readLin(R);

    // ---- carve workspace (floats) ----
    float* base = (float*)d_ws;
    size_t off = 0;
    auto carve = [&](size_t n) { float* p = base + off; off += n; return p; };
    WsBuf w;
    w.e36   = carve((size_t)Ee * DEIN);
    w.e_a   = carve((size_t)Ee * DE);
    w.e_b   = carve((size_t)Ee * DE);
    w.etmp  = carve((size_t)Ee * DE);
    w.ebuf  = carve((size_t)Ee * DE);       // persistent edge state
    w.mu    = carve((size_t)Ee);
    w.rstd  = carve((size_t)Ee);
    w.n_a   = carve((size_t)Nn * 192);
    w.n_b   = carve((size_t)Nn * 192);
    w.ntmp  = carve((size_t)Nn * 192);
    w.hbuf  = carve((size_t)Nn * 192);
    w.xbuf  = carve((size_t)Nn * 192);      // persistent node state
    w.idxb  = carve((size_t)Nn * DI);
    w.njoff = (int*)carve((size_t)Ee);

    // ---- node featurizer: conv -> res -> res -> ln+relu ----
    conv1d_kernel<<<dim3(gblk((long long)Nn * DN)), 256, 0, stream>>>(
        node, conv_w, conv_b, w.n_a, 6, DN);
    run_res(stream, nf_r1, w.n_a, w.n_b, w.ntmp, w.mu, w.rstd, Nn);
    run_res(stream, nf_r2, w.n_b, w.n_a, w.ntmp, w.mu, w.rstd, Nn);
    launch_lnrelu(stream, w.n_a, nf_ln, w.xbuf, Nn, DN);

    // ---- edge featurizer: gather -> res -> res -> ln+relu ----
    gather_rows_kernel<<<dim3(gblk(Ee * DEIN)), 256, 0, stream>>>(
        edgemat, adjmat, w.e36, Ee, DEIN);
    run_res(stream, ef_r1, w.e36, w.e_a, w.etmp, w.mu, w.rstd, (int)Ee);
    run_res(stream, ef_r2, w.e_a, w.e_b, w.etmp, w.mu, w.rstd, (int)Ee);
    launch_lnrelu(stream, w.e_b, ef_ln, w.ebuf, (int)Ee, DE);

    // ---- encoder: 4 RGC layers (dn = 128) ----
    for (int l = 0; l < 4; ++l) run_rgc(stream, enc[l], DN, adjmat, w);

    // ---- masked-residue embedding branch ----
    gather_rows_kernel<<<dim3(gblk((long long)Nn * DI)), 256, 0, stream>>>(
        emb, masked, w.idxb, Nn, DI);
    run_res(stream, idx_r1, w.idxb, w.n_a, w.ntmp, w.mu, w.rstd, Nn);
    launch_lnrelu(stream, w.n_a, idx_ln, w.idxb, Nn, DI);

    // x = concat(x, idx)  -> node state becomes 160-dim
    concat2_kernel<<<dim3(gblk((long long)Nn * (DN + DI))), 256, 0, stream>>>(
        w.xbuf, DN, w.idxb, DI, w.hbuf, Nn);
    copy_kernel<<<dim3(gblk((long long)Nn * (DN + DI))), 256, 0, stream>>>(
        w.hbuf, w.xbuf, (long long)Nn * (DN + DI));

    // ---- decoder: 4 RGC layers (dn = 160) ----
    for (int l = 0; l < 4; ++l) run_rgc(stream, dec[l], DN + DI, adjmat, w);

    // ---- output head: (N x 160) @ (160 x 20) + b ----
    launch_gemm(stream, w.xbuf, Nn, DN + DI, DOUTv, outp, nullptr,
                nullptr, nullptr, nullptr, false, (float*)d_out, nullptr);
}
